// Synthesis_net_38972533244017
// MI455X (gfx1250) — matrix operand order repaired and checked by reading, not compiled, not run on hardware
//
#include <hip/hip_runtime.h>

// CDNA5 / gfx1250, wave32. Matrix math via v_wmma_f32_16x16x32_f16.
typedef _Float16 v16h __attribute__((ext_vector_type(16)));
typedef float    v8f  __attribute__((ext_vector_type(8)));

// -------------------------------------------------------------------------
// Stride-2 transposed conv, k=5, pad=2, out_pad=1 decomposes into 4 phase
// convs. For output phase (py,px): taps ky = 2*ty+py (ty in 0..(py?1:2)),
// input row iy = i + ty - (py?0:1); same in x. Tap counts: 9/6/6/4 (sum 25).
// Phase-major global tap index gt: starts {0,9,15,21} for p = py*2+px.
//
// Channel <-> hardware-K permutation: WMMA only needs A and B to agree on
// which channel occupies each hw-K slot. We choose
//   perm: hwK 0..7 -> ch 0..7, hwK 8..15 -> ch 16..23,
//         hwK 16..23 -> ch 8..15, hwK 24..31 -> ch 24..31
// so that the A fragment per lane is ONE contiguous 32B load from NHWC
// activations (lanes 0-15: ch kb*32+0..15, lanes 16-31: ch kb*32+16..31),
// in natural element order. The permutation is folded into weight packing.
// -------------------------------------------------------------------------

// ---- prep: x fp32 NCHW [4,320,16,16] -> f16 NHWC ------------------------
__global__ void cvt_x_kernel(const float* __restrict__ x, _Float16* __restrict__ xh) {
  int idx = blockIdx.x * 256 + threadIdx.x;
  if (idx >= 4 * 320 * 16 * 16) return;
  int w = idx & 15;
  int h = (idx >> 4) & 15;
  int c = (idx >> 8) % 320;
  int n = idx / (320 * 256);
  xh[((n * 16 + h) * 16 + w) * 320 + c] = (_Float16)x[idx];
}

// ---- prep: stage-1 bias scaled by avrg_s = s1/sn1 -----------------------
__global__ void bias_scale_kernel(const float* __restrict__ b, const float* __restrict__ s,
                                  const float* __restrict__ sn, float* __restrict__ out) {
  int c = threadIdx.x; // 192 threads
  out[c] = b[c] * s[c] / sn[0];
}

// ---- prep: pack OIHW fp32 weights into WMMA B-fragment layout (f16) -----
// Packed index: ((((gt*KB + kb)*numCoT + coT)*32 + lane)*16 + e)
// B 32x16 f16 order used (contiguous k): lane<16 -> N=lane, hwK = e;
//                                  lane>=16 -> N=lane-16, hwK = 16+e.
// Channel ci = kb*32 + hwK, same order as the A loads. All divisors compile-time.
// MODE 0: none; 1: w *= s[co]/sn (stage1, out-ch); 2: w *= sn/s[ci] (stage2, in-ch)
template <int CIN, int COUT_, int NUMCOT, int MODE>
__global__ void pack_w_kernel(const float* __restrict__ W, const float* __restrict__ s,
                              const float* __restrict__ sn, _Float16* __restrict__ wpk) {
  constexpr int KB = CIN / 32;
  constexpr int TOTAL = 25 * KB * NUMCOT * 512;
  int idx = blockIdx.x * 256 + threadIdx.x;
  if (idx >= TOTAL) return;
  int e    = idx & 15;
  int lane = (idx >> 4) & 31;
  int rest = idx >> 9;
  int coT  = rest % NUMCOT; rest /= NUMCOT;
  int kb   = rest % KB;
  int gt   = rest / KB;
  int p, t;
  if      (gt < 9)  { p = 0; t = gt;      }
  else if (gt < 15) { p = 1; t = gt - 9;  }
  else if (gt < 21) { p = 2; t = gt - 15; }
  else              { p = 3; t = gt - 21; }
  int py = p >> 1, px = p & 1;
  int txc = (px == 0) ? 3 : 2;
  int ty = (txc == 3) ? (t / 3) : (t >> 1);
  int tx = (txc == 3) ? (t % 3) : (t & 1);
  int ky = 2 * ty + py;
  int kx = 2 * tx + px;
  int co = coT * 16 + (lane & 15);
  int hwK = ((lane < 16) ? 0 : 16) + e;
  int cperm;
  if      (hwK < 8)  cperm = hwK;        // 0..7   -> 0..7
  else if (hwK < 16) cperm = hwK + 8;    // 8..15  -> 16..23
  else if (hwK < 24) cperm = hwK - 8;    // 16..23 -> 8..15
  else               cperm = hwK;        // 24..31 -> 24..31
  int ci = kb * 32 + hwK;
  float w = 0.f;
  if (co < COUT_) {
    w = W[((co * CIN + ci) * 5 + ky) * 5 + kx];
    if (MODE == 1)      w = w * s[co] / sn[0];
    else if (MODE == 2) w = w * sn[0] / s[ci];
  }
  wpk[idx] = (_Float16)w;
}

// ---- main: one wave computes a (16*MT spatial) x (16*NT cout) macro-tile
// of one phase. Register-blocked: MT*NT f32 accumulators; each A fragment
// reused NT times, each B fragment MT times. With the channel permutation,
// every fragment (A and B) is a single aligned 32B v16h load. Tap loop is
// division-free (nested ty/tx with running gt counter).
template <int CIN, int H, int W, int COUT, int MT, int NT, bool RELU, bool FINAL>
__global__ __launch_bounds__(32) void deconv_wmma_kernel(
    const _Float16* __restrict__ act, const _Float16* __restrict__ wpk,
    const float* __restrict__ bias, void* __restrict__ outp) {
  constexpr int KB = CIN / 32;
  constexpr int numCoT = (COUT + 15) / 16;
  constexpr int HW = H * W;
  constexpr int Ho = 2 * H, Wo = 2 * W;

  const int lane = threadIdx.x;
  const int p  = blockIdx.z, py = p >> 1, px = p & 1;
  const int coT0  = blockIdx.y * NT;
  const int sBase = blockIdx.x * (16 * MT);

  const int tyc = (py == 0) ? 3 : 2;
  const int txc = (px == 0) ? 3 : 2;
  const int gtBase = (py == 0) ? ((px == 0) ? 0 : 9) : ((px == 0) ? 15 : 21);
  const int gtEnd  = gtBase + tyc * txc;

  // A-fragment rows handled by this lane (HW, W are powers of two -> shifts)
  int nA[MT], iAr[MT], jAr[MT];
#pragma unroll
  for (int m = 0; m < MT; ++m) {
    const int sA = sBase + m * 16 + (lane & 15);
    nA[m]  = sA / HW;
    iAr[m] = (sA % HW) / W;
    jAr[m] = sA % W;
  }
  const int kHalf = (lane < 16) ? 0 : 16;  // contiguous 16-channel chunk per lane
  const int dy0 = (py == 0) ? -1 : 0;
  const int dx0 = (px == 0) ? -1 : 0;

  v8f acc[MT][NT] = {};
  int gt = gtBase;
  for (int ty = 0; ty < tyc; ++ty) {
    for (int tx = 0; tx < txc; ++tx, ++gt) {
      const _Float16* abase[MT];
      bool inb[MT];
#pragma unroll
      for (int m = 0; m < MT; ++m) {
        const int iy = iAr[m] + ty + dy0;
        const int ix = jAr[m] + tx + dx0;
        inb[m] = ((unsigned)iy < (unsigned)H) && ((unsigned)ix < (unsigned)W);
        abase[m] = act + (((size_t)(nA[m] * H + iy) * W + ix) * CIN) + kHalf;
      }
      const _Float16* bbase =
          wpk + (((size_t)gt * KB) * numCoT + coT0) * 512 + lane * 16;
      if (gt + 1 < gtEnd) {
        // keep the next tap's weight stream warm in GL2 (global_prefetch_b8)
        __builtin_prefetch(bbase + (size_t)KB * numCoT * 512, 0, 0);
      }
#pragma unroll 2
      for (int kb = 0; kb < KB; ++kb) {
        v16h a[MT];
#pragma unroll
        for (int m = 0; m < MT; ++m) {
          a[m] = inb[m] ? *(const v16h*)(abase[m] + kb * 32) : (v16h){};
        }
#pragma unroll
        for (int nt = 0; nt < NT; ++nt) {
          const v16h b = *(const v16h*)(bbase + ((size_t)kb * numCoT + nt) * 512);
#pragma unroll
          for (int m = 0; m < MT; ++m) {
            acc[m][nt] = __builtin_amdgcn_wmma_f32_16x16x32_f16(
                false, a[m], false, b, (short)0, acc[m][nt], false, false);
          }
        }
      }
    }
  }

  // C/D f32 16x16 layout: lane -> col N = lane&15; VGPR r -> row r (+8 if lane>=16)
  const int rowOff = (lane < 16) ? 0 : 8;
#pragma unroll
  for (int nt = 0; nt < NT; ++nt) {
    const int co = (coT0 + nt) * 16 + (lane & 15);
    const float bv = (co < COUT) ? bias[co] : 0.f;
#pragma unroll
    for (int m = 0; m < MT; ++m) {
#pragma unroll
      for (int r = 0; r < 8; ++r) {
        const int s = sBase + m * 16 + r + rowOff;
        const int n = s / HW;
        const int i = (s % HW) / W;
        const int j = s % W;
        const int oy = 2 * i + py, ox = 2 * j + px;
        float v = acc[m][nt][r] + bv;
        if (RELU) v = (v > 0.f) ? v : 0.f;
        if (FINAL) {
          if (co < COUT)
            ((float*)outp)[(((size_t)n * COUT + co) * Ho + oy) * Wo + ox] = v;
        } else {
          ((_Float16*)outp)[(((size_t)n * Ho + oy) * Wo + ox) * COUT + co] = (_Float16)v;
        }
      }
    }
  }
}

extern "C" void kernel_launch(void* const* d_in, const int* in_sizes, int n_in,
                              void* d_out, int out_size, void* d_ws, size_t ws_size,
                              hipStream_t stream) {
  const float* x   = (const float*)d_in[0];
  const float* W1  = (const float*)d_in[1];
  const float* b1  = (const float*)d_in[2];
  const float* W2  = (const float*)d_in[3];
  const float* b2  = (const float*)d_in[4];
  const float* W3  = (const float*)d_in[5];
  const float* b3  = (const float*)d_in[6];
  const float* W4  = (const float*)d_in[7];
  const float* b4  = (const float*)d_in[8];
  const float* s1  = (const float*)d_in[9];
  const float* sn1 = (const float*)d_in[10];
  (void)in_sizes; (void)n_in; (void)out_size; (void)ws_size;

  // Workspace carve-up (~41 MB total; all stages stay L2-resident)
  char* ws = (char*)d_ws;
  size_t off = 0;
  auto carve = [&](size_t bytes) -> char* {
    char* ptr = ws + off;
    off = (off + bytes + 255) & ~(size_t)255;
    return ptr;
  };
  _Float16* xh   = (_Float16*)carve((size_t)4 * 320 * 16 * 16 * 2);   // x as f16 NHWC
  _Float16* wpk1 = (_Float16*)carve((size_t)25 * 10 * 12 * 512 * 2);  // stage1 packed W
  _Float16* wpk2 = (_Float16*)carve((size_t)25 * 6 * 12 * 512 * 2);
  _Float16* wpk3 = (_Float16*)carve((size_t)25 * 6 * 12 * 512 * 2);
  _Float16* wpk4 = (_Float16*)carve((size_t)25 * 6 * 1 * 512 * 2);
  float*    b1s  = (float*)carve(192 * 4);
  _Float16* h1   = (_Float16*)carve((size_t)4 * 192 * 32 * 32 * 2);   // NHWC f16
  _Float16* h2   = (_Float16*)carve((size_t)4 * 192 * 64 * 64 * 2);
  _Float16* h3   = (_Float16*)carve((size_t)4 * 192 * 128 * 128 * 2);

  // prep
  cvt_x_kernel<<<(4 * 320 * 16 * 16 + 255) / 256, 256, 0, stream>>>(x, xh);
  bias_scale_kernel<<<1, 192, 0, stream>>>(b1, s1, sn1, b1s);
  pack_w_kernel<320, 192, 12, 1>
      <<<(25 * 10 * 12 * 512 + 255) / 256, 256, 0, stream>>>(W1, s1, sn1, wpk1);
  pack_w_kernel<192, 192, 12, 2>
      <<<(25 * 6 * 12 * 512 + 255) / 256, 256, 0, stream>>>(W2, s1, sn1, wpk2);
  pack_w_kernel<192, 192, 12, 0>
      <<<(25 * 6 * 12 * 512 + 255) / 256, 256, 0, stream>>>(W3, s1, sn1, wpk3);
  pack_w_kernel<192, 3, 1, 0>
      <<<(25 * 6 * 1 * 512 + 255) / 256, 256, 0, stream>>>(W4, s1, sn1, wpk4);

  // stages: grid = (spatialTiles = 4*H*W/(16*MT), numCoT/NT, 4 phases), 1 wave/block
  deconv_wmma_kernel<320, 16, 16, 192, 2, 4, true, false>
      <<<dim3(32, 3, 4), 32, 0, stream>>>(xh, wpk1, b1s, h1);
  deconv_wmma_kernel<192, 32, 32, 192, 2, 4, true, false>
      <<<dim3(128, 3, 4), 32, 0, stream>>>(h1, wpk2, b2, h2);
  deconv_wmma_kernel<192, 64, 64, 192, 2, 4, true, false>
      <<<dim3(512, 3, 4), 32, 0, stream>>>(h2, wpk3, b3, h3);
  deconv_wmma_kernel<192, 128, 128, 3, 4, 1, false, true>
      <<<dim3(1024, 1, 4), 32, 0, stream>>>(h3, wpk4, b4, d_out);
}
